// EGNNLayer_54872502174380
// MI455X (gfx1250) — matrix operand order repaired and checked by reading, not compiled, not run on hardware
//
#include <hip/hip_runtime.h>
#include <hip/hip_bf16.h>
#include <math.h>

// ---------- types ----------
typedef __attribute__((ext_vector_type(4)))  unsigned int v4u;
typedef __attribute__((ext_vector_type(8)))  unsigned int v8u;
typedef __attribute__((ext_vector_type(16))) __bf16      v16bf;
typedef __attribute__((ext_vector_type(8)))  float       v8f;

union BF16Frag { v16bf bf; v4u u4[2]; v8u u8; };

__device__ __forceinline__ unsigned short f2bf(float f) {
  unsigned int u = __float_as_uint(f);
  u += 0x7FFFu + ((u >> 16) & 1u);          // round-to-nearest-even
  return (unsigned short)(u >> 16);
}
// fast silu: x * rcp(1+exp(-x)) -- v_exp_f32 + v_add + v_rcp_f32 + v_mul
__device__ __forceinline__ float silu(float x) {
  return x * __builtin_amdgcn_rcpf(1.0f + __expf(-x));
}

#define NB    256      // nodes per batch
#define HID   256      // hidden_nf
#define INF   128      // input_nf / output_nf
#define JB    64       // j rows per block
#define PI_F  3.14159265358979323846f

// ---------- precompute: pe(rel) @ We1[256:384,:]  (511 x 256 table) ----------
__global__ void pe_table_kernel(const float* __restrict__ We1, float* __restrict__ peW) {
  int idx = blockIdx.x;            // 0..510  -> rel = idx-255
  int c   = threadIdx.x;           // 0..255
  float rel = (float)(idx - 255);
  const float LN256_OVER64 = 5.5451774444795624f / 64.0f;  // inv_freq = 256^{-d/64}
  float acc = 0.f;
  for (int d = 0; d < 64; ++d) {
    float invf = __expf(-(float)d * LN256_OVER64);
    float ang  = rel * PI_F * invf;
    float s, co;
    __sincosf(ang, &s, &co);
    acc += s  * We1[(256 + d) * HID + c];   // sin rows
    acc += co * We1[(320 + d) * HID + c];   // cos rows
  }
  peW[idx * HID + c] = acc;
}

// ---------- precompute: f @ We1[0:128] and f @ We1[128:256] per node ----------
__global__ void node_w_kernel(const float* __restrict__ nf, const float* __restrict__ We1,
                              float* __restrict__ fiW, float* __restrict__ fjW) {
  int blk = blockIdx.x;            // b*NB + n, 0..1023
  int c   = threadIdx.x;           // 0..255
  float ai = 0.f, aj = 0.f;
  for (int d = 0; d < INF; ++d) {
    float f = nf[blk * INF + d];
    ai += f * We1[d * HID + c];
    aj += f * We1[(INF + d) * HID + c];
  }
  fiW[blk * HID + c] = ai;
  fjW[blk * HID + c] = aj;
}

// ---------- precompute: pack 256x256 f32 weight into WMMA-B bf16 tiles ----------
// tile (kt,nt): 512 bf16; element order = lane-major, 16 per lane:
//   k = kt*32 + (lane>=16 ? 8 : 0) + (e&7) + (e>=8 ? 16 : 0) ;  n = nt*16 + (lane&15)
__global__ void pack_b_kernel(const float* __restrict__ W, unsigned short* __restrict__ P) {
  int id   = blockIdx.x * 256 + threadIdx.x;   // 65536
  int e    = id & 15;
  int lane = (id >> 4) & 31;
  int tile = id >> 9;
  int nt   = tile & 15;
  int kt   = tile >> 4;
  int k = kt * 32 + ((lane >= 16) ? 8 : 0) + (e & 7) + ((e >> 3) << 4);
  int n = nt * 16 + (lane & 15);
  P[id] = f2bf(W[k * HID + n]);
}

// ---------- main fused kernel: one workgroup per (b,i) ----------
__global__ __launch_bounds__(512)
void egnn_main_kernel(const float* __restrict__ coords, const float* __restrict__ nf,
                      const float* __restrict__ amask, const int* __restrict__ resid,
                      const float* __restrict__ We1, const float* __restrict__ be1,
                      const float* __restrict__ be2,
                      const float* __restrict__ Wn1, const float* __restrict__ bn1,
                      const float* __restrict__ Wn2, const float* __restrict__ bn2,
                      const float* __restrict__ gamma, const float* __restrict__ beta,
                      const float* __restrict__ bc1, const float* __restrict__ Wc2,
                      const float* __restrict__ bc2,
                      const float* __restrict__ fiW, const float* __restrict__ fjW,
                      const float* __restrict__ peW,
                      const unsigned short* __restrict__ pWe2,
                      const unsigned short* __restrict__ pWc1,
                      float* __restrict__ outCoords, float* __restrict__ outH) {
  const int tid = threadIdx.x;
  const int b   = blockIdx.x >> 8;
  const int i   = blockIdx.x & 255;
  const int bN  = b * NB;

  __shared__ __align__(16) unsigned short hidA[JB * HID];  // 32 KB: A tile (hidden1 then msg), bf16
  __shared__ float fiWb[HID], be2s[HID], bc1s[HID], wc2s[HID];
  __shared__ float W3[3 * HID];          // We1 rows 384..386
  __shared__ float aggL[HID];
  __shared__ float dirsL[JB * 3];
  __shared__ float cuL[JB];
  __shared__ int   relL[JB];
  __shared__ float cmi[3];
  __shared__ float cacc[3];
  __shared__ int   residIs;
  __shared__ float nfL[INF], h1L[HID], h2L[INF];
  __shared__ float redS[2];

  // ---- per-workgroup init ----
  if (tid < HID) {
    fiWb[tid] = fiW[(bN + i) * HID + tid] + be1[tid];
    be2s[tid] = be2[tid];
    bc1s[tid] = bc1[tid];
    wc2s[tid] = Wc2[tid];
    aggL[tid] = 0.f;
    W3[tid]           = We1[384 * HID + tid];
    W3[HID + tid]     = We1[385 * HID + tid];
    W3[2 * HID + tid] = We1[386 * HID + tid];
  }
  if (tid < 3) {
    float s = 0.f;
    for (int a = 0; a < 4; ++a) s += coords[((size_t)(bN + i) * 4 + a) * 3 + tid];
    cmi[tid]  = s * 0.25f;
    cacc[tid] = 0.f;
  }
  if (tid == 4) residIs = resid[bN + i];
  if (tid < INF) nfL[tid] = nf[(size_t)(bN + i) * INF + tid];
  const float bc2v = bc2[0];
  __syncthreads();

  // ---- wave tile mapping ----
  const int lane  = tid & 31;
  const int wv    = tid >> 5;
  const int mt    = wv & 3;          // M tile (16 rows) within 64-row block
  const int ng    = wv >> 2;         // N group: 4 n-tiles = 64 cols
  const int rowA  = mt * 16 + (lane & 15);
  const int k8    = (lane >= 16) ? 8 : 0;   // A-fragment K sub-offset
  const int hiRow = (lane >= 16) ? 8 : 0;   // C-fragment row sub-offset

  // per-wave base pointers into packed weights (tile ng*4 .. ng*4+3 of each kt row)
  const unsigned short* pB2base = pWe2 + (size_t)(ng * 4) * 512 + (size_t)lane * 16;
  const unsigned short* pB3base = pWc1 + (size_t)(ng * 4) * 512 + (size_t)lane * 16;

  #pragma clang loop unroll(disable)
  for (int jb = 0; jb < 4; ++jb) {
    const int j0 = jb * JB;

    // per-row dirs / rel table, cu init
    if (tid < JB) {
      int j = j0 + tid;
      const float* cp = &coords[((size_t)(bN + j) * 4) * 3];
      float c0 = 0.f, c1 = 0.f, c2 = 0.f;
      for (int a = 0; a < 4; ++a) { c0 += cp[a*3+0]; c1 += cp[a*3+1]; c2 += cp[a*3+2]; }
      dirsL[tid*3+0] = cmi[0] - c0 * 0.25f;
      dirsL[tid*3+1] = cmi[1] - c1 * 0.25f;
      dirsL[tid*3+2] = cmi[2] - c2 * 0.25f;
      int rl = residIs - resid[bN + j] + 255;
      relL[tid] = rl < 0 ? 0 : (rl > 510 ? 510 : rl);
      cuL[tid] = bc2v;
    }
    __syncthreads();

    // ---- build silu(hidden1) bf16 tile: thread = one column, 32 rows ----
    {
      const int n     = tid & 255;
      const int rbase = (tid >> 8) * 32;
      const float fi = fiWb[n];
      const float w0 = W3[n], w1 = W3[HID + n], w2 = W3[2 * HID + n];
      #pragma clang loop unroll_count(4)
      for (int rr = 0; rr < 32; ++rr) {
        int r = rbase + rr;
        int j = j0 + r;
        float dx = dirsL[r*3+0], dy = dirsL[r*3+1], dz = dirsL[r*3+2];
        float x = fi + fjW[(size_t)(bN + j) * HID + n] + peW[(size_t)relL[r] * HID + n]
                + dx * w0 + dy * w1 + dz * w2;
        hidA[r * HID + n] = f2bf(silu(x));
      }
    }
    __syncthreads();

    // ---- GEMM2: msg_pre = hidden1 @ We2  (M=64,N=256,K=256), wave: mt x 4 n-tiles
    v8f acc[4];
    #pragma unroll
    for (int t = 0; t < 4; ++t)
      #pragma unroll
      for (int v = 0; v < 8; ++v) acc[t][v] = 0.f;

    #pragma clang loop unroll(disable)
    for (int kt = 0; kt < 8; ++kt) {
      BF16Frag a;
      a.u4[0] = *(const v4u*)&hidA[rowA * HID + kt * 32 + k8];
      a.u4[1] = *(const v4u*)&hidA[rowA * HID + kt * 32 + 16 + k8];
      const unsigned short* bp = pB2base + (size_t)kt * (16 * 512);
      #pragma unroll
      for (int t = 0; t < 4; ++t) {
        BF16Frag bfr;
        bfr.u8 = *(const v8u*)(bp + (size_t)t * 512);
        acc[t] = __builtin_amdgcn_wmma_f32_16x16x32_bf16(
            false, a.bf, false, bfr.bf, (short)0, acc[t], false, false);
      }
    }

    // silu(+be2) -> msg in regs; agg += column sums
    #pragma unroll
    for (int t = 0; t < 4; ++t) {
      int colB = ng * 64 + t * 16 + (lane & 15);
      float bsum = 0.f;
      float bv = be2s[colB];
      #pragma unroll
      for (int v = 0; v < 8; ++v) {
        float m = silu(acc[t][v] + bv);
        acc[t][v] = m;
        bsum += m;
      }
      bsum += __shfl_xor(bsum, 16, 32);
      if (lane < 16) atomicAdd(&aggL[colB], bsum);
    }
    __syncthreads();   // all GEMM2 reads of hidA done

    // store msg (bf16) back into hidA as A for GEMM3
    #pragma unroll
    for (int t = 0; t < 4; ++t) {
      int colB = ng * 64 + t * 16 + (lane & 15);
      #pragma unroll
      for (int v = 0; v < 8; ++v) {
        int row = mt * 16 + v + hiRow;
        hidA[row * HID + colB] = f2bf(acc[t][v]);
      }
    }
    __syncthreads();

    // ---- GEMM3: t = msg @ Wc1 ; cu_row = sum_col silu(t+bc1)*wc2 ----
    #pragma unroll
    for (int t = 0; t < 4; ++t)
      #pragma unroll
      for (int v = 0; v < 8; ++v) acc[t][v] = 0.f;

    #pragma clang loop unroll(disable)
    for (int kt = 0; kt < 8; ++kt) {
      BF16Frag a;
      a.u4[0] = *(const v4u*)&hidA[rowA * HID + kt * 32 + k8];
      a.u4[1] = *(const v4u*)&hidA[rowA * HID + kt * 32 + 16 + k8];
      const unsigned short* bp = pB3base + (size_t)kt * (16 * 512);
      #pragma unroll
      for (int t = 0; t < 4; ++t) {
        BF16Frag bfr;
        bfr.u8 = *(const v8u*)(bp + (size_t)t * 512);
        acc[t] = __builtin_amdgcn_wmma_f32_16x16x32_bf16(
            false, a.bf, false, bfr.bf, (short)0, acc[t], false, false);
      }
    }

    float rowpart[8];
    #pragma unroll
    for (int v = 0; v < 8; ++v) rowpart[v] = 0.f;
    #pragma unroll
    for (int t = 0; t < 4; ++t) {
      int colB = ng * 64 + t * 16 + (lane & 15);
      float w  = wc2s[colB];
      float bv = bc1s[colB];
      #pragma unroll
      for (int v = 0; v < 8; ++v)
        rowpart[v] += silu(acc[t][v] + bv) * w;
    }
    #pragma unroll
    for (int v = 0; v < 8; ++v) {
      float s = rowpart[v];
      s += __shfl_xor(s, 1, 32);
      s += __shfl_xor(s, 2, 32);
      s += __shfl_xor(s, 4, 32);
      s += __shfl_xor(s, 8, 32);
      if ((lane & 15) == 0) atomicAdd(&cuL[mt * 16 + v + hiRow], s);
    }
    __syncthreads();

    if (tid < JB) {
      float cv = cuL[tid];
      atomicAdd(&cacc[0], cv * dirsL[tid*3+0]);
      atomicAdd(&cacc[1], cv * dirsL[tid*3+1]);
      atomicAdd(&cacc[2], cv * dirsL[tid*3+2]);
    }
    __syncthreads();
  }

  // ---- node MLP + LayerNorm ----
  if (tid < HID) {
    float a = bn1[tid];
    for (int d = 0; d < INF; ++d) a += nfL[d] * Wn1[d * HID + tid];
    for (int d = 0; d < HID; ++d) a += aggL[d] * Wn1[(INF + d) * HID + tid];
    h1L[tid] = silu(a);
  }
  if (tid == 0) { redS[0] = 0.f; redS[1] = 0.f; }
  __syncthreads();
  if (tid < INF) {
    float a = bn2[tid];
    for (int d = 0; d < HID; ++d) a += h1L[d] * Wn2[d * INF + tid];
    h2L[tid] = a;
    atomicAdd(&redS[0], a);
    atomicAdd(&redS[1], a * a);
  }
  __syncthreads();
  if (tid < INF) {
    float mu  = redS[0] * (1.0f / 128.0f);
    float var = redS[1] * (1.0f / 128.0f) - mu * mu;
    float y = (h2L[tid] - mu) * rsqrtf(var + 1e-5f) * gamma[tid] + beta[tid];
    outH[(size_t)(bN + i) * INF + tid] = y;
  }
  if (tid < 12) {
    int a = tid / 3, d = tid % 3;
    size_t off = ((size_t)(bN + i) * 4 + a) * 3 + d;
    outCoords[off] = coords[off] + cacc[d] * amask[(size_t)(bN + i) * 4 + a];
  }
}

// ---------- launch ----------
extern "C" void kernel_launch(void* const* d_in, const int* in_sizes, int n_in,
                              void* d_out, int out_size, void* d_ws, size_t ws_size,
                              hipStream_t stream) {
  const float* coords = (const float*)d_in[0];
  const float* nf     = (const float*)d_in[1];
  const float* amask  = (const float*)d_in[2];
  const int*   resid  = (const int*)d_in[3];
  const float* We1 = (const float*)d_in[4];
  const float* be1 = (const float*)d_in[5];
  const float* We2 = (const float*)d_in[6];
  const float* be2 = (const float*)d_in[7];
  const float* Wn1 = (const float*)d_in[8];
  const float* bn1 = (const float*)d_in[9];
  const float* Wn2 = (const float*)d_in[10];
  const float* bn2 = (const float*)d_in[11];
  const float* gamma = (const float*)d_in[12];
  const float* beta  = (const float*)d_in[13];
  const float* Wc1 = (const float*)d_in[14];
  const float* bc1 = (const float*)d_in[15];
  const float* Wc2 = (const float*)d_in[16];
  const float* bc2 = (const float*)d_in[17];

  char* ws = (char*)d_ws;
  float* peW = (float*)(ws);                                  // 511*256*4   = 523264 B
  float* fiW = (float*)(ws + 523264);                         // 1024*256*4  = 1 MB
  float* fjW = (float*)(ws + 523264 + 1048576);               // 1 MB
  unsigned short* pWe2 = (unsigned short*)(ws + 523264 + 2097152);           // 128 KB
  unsigned short* pWc1 = (unsigned short*)(ws + 523264 + 2097152 + 131072);  // 128 KB

  pe_table_kernel<<<511, 256, 0, stream>>>(We1, peW);
  node_w_kernel<<<1024, 256, 0, stream>>>(nf, We1, fiW, fjW);
  pack_b_kernel<<<256, 256, 0, stream>>>(We2, pWe2);
  pack_b_kernel<<<256, 256, 0, stream>>>(Wc1, pWc1);

  float* outCoords = (float*)d_out;
  float* outH = outCoords + 4 * 256 * 4 * 3;
  egnn_main_kernel<<<1024, 512, 0, stream>>>(coords, nf, amask, resid,
      We1, be1, be2, Wn1, bn1, Wn2, bn2, gamma, beta, bc1, Wc2, bc2,
      fiW, fjW, peW, pWe2, pWc1, outCoords, outH);
}
